// GNOT_15324443312956
// MI455X (gfx1250) — matrix operand order repaired and checked by reading, not compiled, not run on hardware
//
#include <hip/hip_runtime.h>
#include <math.h>

// ---------------------------------------------------------------------------
// GNOT forward for MI455X (gfx1250, wave32, WMMA).
// All dense GEMMs (K>=32) run through v_wmma_f32_16x16x32_f16 with fp32
// accumulation; weights are converted once per launch to f16 [N][K].
// GEMM: 128x128 block tile, 4 waves x (32 rows x 128 cols), 16 WMMAs per
// K-chunk per wave between barriers; fp32 activations cvt'd to f16 in LDS.
// ---------------------------------------------------------------------------

typedef __attribute__((ext_vector_type(16))) _Float16 v16h;
typedef __attribute__((ext_vector_type(8)))  _Float16 v8h;
typedef __attribute__((ext_vector_type(8)))  float    v8f;

union Frag16 { v16h v; v8h h[2]; };

#define CC    128          // hidden
#define HH    8            // heads
#define EE    3            // experts
#define FFDIM 512          // mlp dim
#define BATCH 4
#define NSEQ  8192
#define MTOK  (BATCH*NSEQ) // 32768 tokens

enum { MODE_STORE=0, MODE_GELU=1, MODE_RESADD=2, MODE_GATE_STORE=3, MODE_GATE_ACC=4 };

__device__ __forceinline__ float gelu_exact(float x) {
    return 0.5f * x * (1.0f + erff(x * 0.70710678118654752440f));
}

// ------------------- weight conversion: f32 [K][N] -> f16 [N][K] -----------
__global__ void convert_wT_kernel(const float* __restrict__ src,
                                  _Float16* __restrict__ dst, int K, int N) {
    int idx = blockIdx.x * 256 + threadIdx.x;
    if (idx >= K * N) return;
    int k = idx / N, n = idx - k * N;
    dst[(size_t)n * K + k] = (_Float16)src[idx];
}

// ------------------- WMMA GEMM ---------------------------------------------
// Y[M,N] fp32 = epilogue( X[M,K] fp32 * W(f16,[N][K]) + bias[N] )
// 128 threads = 4 waves; block tile 128x128; wave tile 32 rows x 128 cols.
// Requires M % 128 == 0, N % 128 == 0, K % 32 == 0.
__global__ __launch_bounds__(128)
void gemm_wmma_kernel(const float* __restrict__ X, const _Float16* __restrict__ Wt,
                      const float* __restrict__ bias, float* __restrict__ Y,
                      const float* __restrict__ R, const float* __restrict__ gate,
                      int gidx, int N, int K, int mode)
{
    // rows padded to 40 halves (80B) -> bank-conflict-free b128 fragment reads
    __shared__ _Float16 lsA[128 * 40];   // 128 activation rows x 32 K
    __shared__ _Float16 lsW[128 * 40];   // 128 output cols   x 32 K

    const int tid  = threadIdx.x;
    const int wave = tid >> 5;
    const int lane = tid & 31;
    const int g    = lane >> 4;     // half-wave
    const int nr   = lane & 15;
    const int m0   = blockIdx.x * 128;
    const int n0   = blockIdx.y * 128;

    v8f acc[16] = {};   // [j=0..7 col tiles][r=0..1 row tiles] -> acc[j*2+r]

    for (int k0 = 0; k0 < K; k0 += 32) {
        // stage X row `tid` (32 floats -> f16)
        {
            const float* src = X + (size_t)(m0 + tid) * K + k0;
            _Float16* dst = &lsA[tid * 40];
            #pragma unroll
            for (int p = 0; p < 4; ++p) {
                float4 fa = *(const float4*)(src + p * 8 + 0);
                float4 fb = *(const float4*)(src + p * 8 + 4);
                v8h hh;
                hh[0]=(_Float16)fa.x; hh[1]=(_Float16)fa.y; hh[2]=(_Float16)fa.z; hh[3]=(_Float16)fa.w;
                hh[4]=(_Float16)fb.x; hh[5]=(_Float16)fb.y; hh[6]=(_Float16)fb.z; hh[7]=(_Float16)fb.w;
                *(v8h*)(dst + p * 8) = hh;
            }
        }
        // stage W row `tid` (32 halves, already f16 K-major)
        {
            const _Float16* src = Wt + (size_t)(n0 + tid) * K + k0;
            _Float16* dst = &lsW[tid * 40];
            #pragma unroll
            for (int p = 0; p < 4; ++p)
                *(v8h*)(dst + p * 8) = *(const v8h*)(src + p * 8);
        }
        __syncthreads();

        // A fragments: rows (wave*32 + r*16 + lane%16); ISA 16-bit A layout
        Frag16 a0, a1;
        {
            const _Float16* ab = &lsA[(wave * 32 + nr) * 40];
            a0.h[0] = *(const v8h*)(ab + g * 8);         // K = g*8 .. +7
            a0.h[1] = *(const v8h*)(ab + 16 + g * 8);    // K = 16+g*8 .. +7
            ab += 16 * 40;
            a1.h[0] = *(const v8h*)(ab + g * 8);
            a1.h[1] = *(const v8h*)(ab + 16 + g * 8);
        }
        #pragma unroll
        for (int j = 0; j < 8; ++j) {
            // B fragment: col = lane%16 of tile j, K = g*8 + 0..7, 16+g*8 + 0..7
            Frag16 bf;
            const _Float16* bbase = &lsW[(j * 16 + nr) * 40 + g * 8];
            bf.h[0] = *(const v8h*)(bbase);
            bf.h[1] = *(const v8h*)(bbase + 16);
            acc[j*2+0] = __builtin_amdgcn_wmma_f32_16x16x32_f16(
                false, a0.v, false, bf.v, (short)0, acc[j*2+0], false, false);
            acc[j*2+1] = __builtin_amdgcn_wmma_f32_16x16x32_f16(
                false, a1.v, false, bf.v, (short)0, acc[j*2+1], false, false);
        }
        __syncthreads();
    }

    #pragma unroll
    for (int j = 0; j < 8; ++j) {
        const int col = n0 + j * 16 + nr;
        const float bv = bias[col];
        #pragma unroll
        for (int r = 0; r < 2; ++r) {
            const int mrow = m0 + wave * 32 + r * 16 + g * 8;  // C/D: VGPR i -> row g*8+i
            #pragma unroll
            for (int i = 0; i < 8; ++i) {
                const int row = mrow + i;
                const size_t off = (size_t)row * N + col;
                float val = acc[j*2+r][i] + bv;
                if (mode == MODE_GELU)            Y[off] = gelu_exact(val);
                else if (mode == MODE_RESADD)     Y[off] = R[off] + val;
                else if (mode == MODE_GATE_STORE) Y[off] = gate[(size_t)row * EE + gidx] * val;
                else if (mode == MODE_GATE_ACC)   Y[off] += gate[(size_t)row * EE + gidx] * val;
                else                              Y[off] = val;
            }
        }
    }
}

// ------------------- pre-net front: gelu(lin_pre(concat(pos,fx))) ----------
__global__ void prep_front_kernel(const float* __restrict__ pos, const float* __restrict__ fx,
                                  const float* __restrict__ w, const float* __restrict__ b,
                                  float* __restrict__ out) {
    int idx = blockIdx.x * 256 + threadIdx.x;     // m*256 + j, exact grid
    int m = idx >> 8, j = idx & 255;
    float z = pos[m*2] * w[j] + pos[m*2+1] * w[256 + j] + fx[m] * w[512 + j] + b[j];
    out[idx] = gelu_exact(z);
}

// ------------------- gate front: gelu(lin0(pos)) ---------------------------
__global__ void gate_front_kernel(const float* __restrict__ pos, const float* __restrict__ w,
                                  const float* __restrict__ b, float* __restrict__ out) {
    int idx = blockIdx.x * 256 + threadIdx.x;     // m*512 + f, exact grid
    int m = idx >> 9, f = idx & 511;
    float z = pos[m*2] * w[f] + pos[m*2+1] * w[512 + f] + b[f];
    out[idx] = gelu_exact(z);
}

// ------------------- gate tail: lin2 (512->3) + softmax, wave per token ----
__global__ __launch_bounds__(128)
void gate_final_kernel(const float* __restrict__ hid, const float* __restrict__ w,
                       const float* __restrict__ b, float* __restrict__ gate) {
    int m = blockIdx.x * 4 + (threadIdx.x >> 5);
    int lane = threadIdx.x & 31;
    const float* hr = hid + (size_t)m * FFDIM;
    float s0 = 0.f, s1 = 0.f, s2 = 0.f;
    for (int f = lane; f < FFDIM; f += 32) {
        float hv = hr[f];
        s0 += hv * w[f*3+0]; s1 += hv * w[f*3+1]; s2 += hv * w[f*3+2];
    }
    for (int off = 16; off > 0; off >>= 1) {
        s0 += __shfl_xor(s0, off, 32);
        s1 += __shfl_xor(s1, off, 32);
        s2 += __shfl_xor(s2, off, 32);
    }
    if (lane == 0) {
        float g0 = s0 + b[0], g1 = s1 + b[1], g2 = s2 + b[2];
        float mx = fmaxf(g0, fmaxf(g1, g2));
        float e0 = expf(g0 - mx), e1 = expf(g1 - mx), e2 = expf(g2 - mx);
        float inv = 1.0f / (e0 + e1 + e2);
        gate[(size_t)m*3+0] = e0*inv; gate[(size_t)m*3+1] = e1*inv; gate[(size_t)m*3+2] = e2*inv;
    }
}

// ------------------- LayerNorm, wave per token; optional residual ----------
__global__ __launch_bounds__(128)
void lnorm_kernel(const float* __restrict__ src, const float* __restrict__ gg,
                  const float* __restrict__ bb, const float* __restrict__ res,
                  float* __restrict__ dst) {
    int m = blockIdx.x * 4 + (threadIdx.x >> 5);
    int lane = threadIdx.x & 31;
    const float* x = src + (size_t)m * CC;
    float v0 = x[lane], v1 = x[lane+32], v2 = x[lane+64], v3 = x[lane+96];
    float s = v0 + v1 + v2 + v3;
    for (int off = 16; off > 0; off >>= 1) s += __shfl_xor(s, off, 32);
    float mean = s * (1.0f / CC);
    float d0 = v0-mean, d1 = v1-mean, d2 = v2-mean, d3 = v3-mean;
    float q = d0*d0 + d1*d1 + d2*d2 + d3*d3;
    for (int off = 16; off > 0; off >>= 1) q += __shfl_xor(q, off, 32);
    float rstd = rsqrtf(q * (1.0f / CC) + 1e-5f);
    float* y = dst + (size_t)m * CC;
    const float* r = res ? res + (size_t)m * CC : nullptr;
    float o0 = d0*rstd*gg[lane]    + bb[lane];
    float o1 = d1*rstd*gg[lane+32] + bb[lane+32];
    float o2 = d2*rstd*gg[lane+64] + bb[lane+64];
    float o3 = d3*rstd*gg[lane+96] + bb[lane+96];
    if (r) { o0 += r[lane]; o1 += r[lane+32]; o2 += r[lane+64]; o3 += r[lane+96]; }
    y[lane] = o0; y[lane+32] = o1; y[lane+64] = o2; y[lane+96] = o3;
}

// ------------------- per-(token,head) 16-wide softmax (q and k) ------------
__global__ void softmax16_kernel(float* __restrict__ buf) {
    int id = blockIdx.x * 256 + threadIdx.x;   // exact grid: MTOK*HH threads
    float* p = buf + (size_t)id * 16;
    float v[16]; float mx = -1e30f;
    #pragma unroll
    for (int i = 0; i < 16; ++i) { v[i] = p[i]; mx = fmaxf(mx, v[i]); }
    float s = 0.f;
    #pragma unroll
    for (int i = 0; i < 16; ++i) { v[i] = expf(v[i] - mx); s += v[i]; }
    float inv = 1.0f / s;
    #pragma unroll
    for (int i = 0; i < 16; ++i) p[i] = v[i] * inv;
}

// ------------------- ctx[d][e] = sum_t k[t,d]*v[t,e]; ksum[d] --------------
// one block per (b,h); 256 threads = (d,e) pairs; k/v staged in LDS
__global__ __launch_bounds__(256)
void ctx_kernel(const float* __restrict__ k, const float* __restrict__ v,
                float* __restrict__ ctxb) {
    __shared__ float lsK[64 * 16];
    __shared__ float lsV[64 * 16];
    int bh = blockIdx.x;
    int b = bh >> 3, h = bh & 7;
    int tid = threadIdx.x;
    int d = tid >> 4, e = tid & 15;
    float acc = 0.f, ks = 0.f;
    for (int t0 = 0; t0 < NSEQ; t0 += 64) {
        for (int i = tid; i < 64 * 32; i += 256) {
            int tok = i >> 5, c = i & 31;
            size_t base = ((size_t)(b * NSEQ + t0 + tok)) * CC + h * 16;
            if (c < 16) lsK[tok*16 + c]      = k[base + c];
            else        lsV[tok*16 + (c-16)] = v[base + (c - 16)];
        }
        __syncthreads();
        #pragma unroll 4
        for (int tok = 0; tok < 64; ++tok) {
            float kv = lsK[tok*16 + d];
            acc += kv * lsV[tok*16 + e];
            ks  += kv;
        }
        __syncthreads();
    }
    ctxb[(size_t)bh * 272 + d*16 + e] = acc;
    if (e == 0) ctxb[(size_t)bh * 272 + 256 + d] = ks;
}

// ------------------- o = (q @ ctx) * dinv + q ------------------------------
__global__ void attn_out_kernel(const float* __restrict__ q, const float* __restrict__ ctxb,
                                float* __restrict__ o) {
    int id = blockIdx.x * 256 + threadIdx.x;   // exact: MTOK*HH
    int m = id >> 3, h = id & 7;
    int bh = (m >> 13) * 8 + h;                 // batch = m / NSEQ
    const float* ctx  = ctxb + (size_t)bh * 272;
    const float* ksum = ctx + 256;
    const float* qr = q + (size_t)m * CC + h * 16;
    float qa[16], dsum = 0.f;
    #pragma unroll
    for (int d = 0; d < 16; ++d) { qa[d] = qr[d]; dsum += qa[d] * ksum[d]; }
    float dinv = 1.0f / dsum;
    float out[16];
    #pragma unroll
    for (int e = 0; e < 16; ++e) out[e] = 0.f;
    #pragma unroll
    for (int d = 0; d < 16; ++d) {
        float qv = qa[d];
        #pragma unroll
        for (int e = 0; e < 16; ++e) out[e] += qv * ctx[d*16 + e];
    }
    float* orow = o + (size_t)m * CC + h * 16;
    #pragma unroll
    for (int e = 0; e < 16; ++e) orow[e] = out[e] * dinv + qa[e];
}

// ------------------- final projection 128 -> 1, wave per token -------------
__global__ __launch_bounds__(128)
void out_kernel(const float* __restrict__ h, const float* __restrict__ w,
                const float* __restrict__ b, float* __restrict__ y) {
    int m = blockIdx.x * 4 + (threadIdx.x >> 5);
    int lane = threadIdx.x & 31;
    const float* x = h + (size_t)m * CC;
    float s = x[lane]*w[lane] + x[lane+32]*w[lane+32]
            + x[lane+64]*w[lane+64] + x[lane+96]*w[lane+96];
    for (int off = 16; off > 0; off >>= 1) s += __shfl_xor(s, off, 32);
    if (lane == 0) y[m] = s + b[0];
}

// ===========================================================================
extern "C" void kernel_launch(void* const* d_in, const int* in_sizes, int n_in,
                              void* d_out, int out_size, void* d_ws, size_t ws_size,
                              hipStream_t stream)
{
    (void)in_sizes; (void)n_in; (void)out_size; (void)ws_size;
    const float* pos = (const float*)d_in[0];   // [B,N,2]
    const float* fx  = (const float*)d_in[1];   // [B,N,1]
    // Param leaves in jax tree_flatten (sorted-dict) order: blocks[0..3] (40 each), out, pre
    const float* const* PL = (const float* const*)(d_in + 2);

    const size_t MB = 1ull << 20;
    char* ws = (char*)d_ws;
    float* h    = (float*)(ws + 0*MB);     // [32768,128] activations
    float* t0   = (float*)(ws + 16*MB);
    float* t1   = (float*)(ws + 32*MB);
    float* qb   = (float*)(ws + 48*MB);
    float* kb   = (float*)(ws + 64*MB);
    float* vb   = (float*)(ws + 80*MB);
    float* ob   = (float*)(ws + 96*MB);
    float* hidA = (float*)(ws + 112*MB);   // [32768,512] (also pre [32768,256])
    float* hidB = (float*)(ws + 176*MB);   // [32768,512]
    float* gate = (float*)(ws + 240*MB);   // [32768,3]
    float* ctxb = (float*)(ws + 241*MB);   // 32 * (256+16)
    _Float16* warena = (_Float16*)(ws + 242*MB);
    size_t woff = 0;

    auto conv = [&](const float* src, int K, int N) -> const _Float16* {
        _Float16* dst = warena + woff;
        woff += (size_t)K * N;
        int tot = K * N;
        convert_wT_kernel<<<(tot + 255)/256, 256, 0, stream>>>(src, dst, K, N);
        return dst;
    };
    auto gemm = [&](const float* X, const _Float16* Wt, const float* bias, float* Y,
                    const float* R, const float* gt, int gidx, int N, int K, int mode) {
        gemm_wmma_kernel<<<dim3(MTOK/128, N/128), 128, 0, stream>>>(
            X, Wt, bias, Y, R, gt, gidx, N, K, mode);
    };

    // --------------- pre-net: h = lin_post(gelu(lin_pre([pos,fx]))) --------
    const float* post_b = PL[162]; const float* post_w = PL[163]; // (256,128)
    const float* pre_b  = PL[164]; const float* pre_w  = PL[165]; // (3,256)
    const _Float16* post_w16 = conv(post_w, 2*CC, CC);
    prep_front_kernel<<<MTOK, 256, 0, stream>>>(pos, fx, pre_w, pre_b, hidA);
    gemm(hidA, post_w16, post_b, h, nullptr, nullptr, 0, CC, 2*CC, MODE_STORE);

    for (int blk = 0; blk < 4; ++blk) {
        const float* const* L = PL + 40 * blk;
        // leaves per block (sorted keys): ca{k,proj,q,v}{b,w}, gate[0..2]{b,w},
        // ln1..ln5{b,g}, moe1{b1,b2,w1,w2}, moe2{...}, sa{k,proj,q,v}{b,w}
        const float *ca_kb=L[0], *ca_pb=L[2], *ca_qb=L[4], *ca_vb=L[6];
        const _Float16 *ca_kw = conv(L[1], CC, CC);
        const _Float16 *ca_pw = conv(L[3], CC, CC);
        const _Float16 *ca_qw = conv(L[5], CC, CC);
        const _Float16 *ca_vw = conv(L[7], CC, CC);
        const float *g0b=L[8], *g0w=L[9], *g1b=L[10], *g2b=L[12], *g2w=L[13];
        const _Float16 *g1w = conv(L[11], FFDIM, FFDIM);
        const float *ln1b=L[14], *ln1g=L[15], *ln2b=L[16], *ln2g=L[17];
        const float *ln3b=L[18], *ln3g=L[19], *ln4b=L[20], *ln4g=L[21];
        const float *ln5b=L[22], *ln5g=L[23];
        const float *m1b1=L[24], *m1b2=L[25], *m2b1=L[28], *m2b2=L[29];
        const _Float16 *m1w1[EE], *m1w2[EE], *m2w1[EE], *m2w2[EE];
        for (int e = 0; e < EE; ++e) m1w1[e] = conv(L[26] + (size_t)e*CC*FFDIM, CC, FFDIM);
        for (int e = 0; e < EE; ++e) m1w2[e] = conv(L[27] + (size_t)e*FFDIM*CC, FFDIM, CC);
        for (int e = 0; e < EE; ++e) m2w1[e] = conv(L[30] + (size_t)e*CC*FFDIM, CC, FFDIM);
        for (int e = 0; e < EE; ++e) m2w2[e] = conv(L[31] + (size_t)e*FFDIM*CC, FFDIM, CC);
        const float *sa_kb=L[32], *sa_pb=L[34], *sa_qb=L[36], *sa_vb=L[38];
        const _Float16 *sa_kw = conv(L[33], CC, CC);
        const _Float16 *sa_pw = conv(L[35], CC, CC);
        const _Float16 *sa_qw = conv(L[37], CC, CC);
        const _Float16 *sa_vw = conv(L[39], CC, CC);

        // -------- gate MLP: softmax(lin2(gelu(lin1(gelu(lin0(pos)))))) -----
        gate_front_kernel<<<MTOK*FFDIM/256, 256, 0, stream>>>(pos, g0w, g0b, hidA);
        gemm(hidA, g1w, g1b, hidB, nullptr, nullptr, 0, FFDIM, FFDIM, MODE_GELU);
        gate_final_kernel<<<MTOK/4, 128, 0, stream>>>(hidB, g2w, g2b, gate);

        // -------- cross attention: x += attn(ln1(x), ln2(x)) ---------------
        lnorm_kernel<<<MTOK/4, 128, 0, stream>>>(h, ln1g, ln1b, nullptr, t0);
        lnorm_kernel<<<MTOK/4, 128, 0, stream>>>(h, ln2g, ln2b, nullptr, t1);
        gemm(t0, ca_qw, ca_qb, qb, nullptr, nullptr, 0, CC, CC, MODE_STORE);
        gemm(t1, ca_kw, ca_kb, kb, nullptr, nullptr, 0, CC, CC, MODE_STORE);
        gemm(t1, ca_vw, ca_vb, vb, nullptr, nullptr, 0, CC, CC, MODE_STORE);
        softmax16_kernel<<<MTOK*HH/256, 256, 0, stream>>>(qb);
        softmax16_kernel<<<MTOK*HH/256, 256, 0, stream>>>(kb);
        ctx_kernel<<<BATCH*HH, 256, 0, stream>>>(kb, vb, ctxb);
        attn_out_kernel<<<MTOK*HH/256, 256, 0, stream>>>(qb, ctxb, ob);
        gemm(ob, ca_pw, ca_pb, h, h, nullptr, 0, CC, CC, MODE_RESADD);

        // -------- moe1 then x += ln3(moe1_out) -----------------------------
        for (int e = 0; e < EE; ++e) {
            gemm(h, m1w1[e], m1b1 + e*FFDIM, hidB, nullptr, nullptr, 0, FFDIM, CC, MODE_GELU);
            gemm(hidB, m1w2[e], m1b2 + e*CC, ob, nullptr, gate, e, CC, FFDIM,
                 e == 0 ? MODE_GATE_STORE : MODE_GATE_ACC);
        }
        lnorm_kernel<<<MTOK/4, 128, 0, stream>>>(ob, ln3g, ln3b, h, h);

        // -------- self attention: xs = ln4(x); x += attn(xs, xs) -----------
        lnorm_kernel<<<MTOK/4, 128, 0, stream>>>(h, ln4g, ln4b, nullptr, t0);
        gemm(t0, sa_qw, sa_qb, qb, nullptr, nullptr, 0, CC, CC, MODE_STORE);
        gemm(t0, sa_kw, sa_kb, kb, nullptr, nullptr, 0, CC, CC, MODE_STORE);
        gemm(t0, sa_vw, sa_vb, vb, nullptr, nullptr, 0, CC, CC, MODE_STORE);
        softmax16_kernel<<<MTOK*HH/256, 256, 0, stream>>>(qb);
        softmax16_kernel<<<MTOK*HH/256, 256, 0, stream>>>(kb);
        ctx_kernel<<<BATCH*HH, 256, 0, stream>>>(kb, vb, ctxb);
        attn_out_kernel<<<MTOK*HH/256, 256, 0, stream>>>(qb, ctxb, ob);
        gemm(ob, sa_pw, sa_pb, h, h, nullptr, 0, CC, CC, MODE_RESADD);

        // -------- moe2 then x += ln5(moe2_out) -----------------------------
        for (int e = 0; e < EE; ++e) {
            gemm(h, m2w1[e], m2b1 + e*FFDIM, hidB, nullptr, nullptr, 0, FFDIM, CC, MODE_GELU);
            gemm(hidB, m2w2[e], m2b2 + e*CC, ob, nullptr, gate, e, CC, FFDIM,
                 e == 0 ? MODE_GATE_STORE : MODE_GATE_ACC);
        }
        lnorm_kernel<<<MTOK/4, 128, 0, stream>>>(ob, ln5g, ln5b, h, h);
    }

    // --------------- output head 128 -> 1 ----------------------------------
    const float* out_b = PL[160]; const float* out_w = PL[161];
    out_kernel<<<MTOK/4, 128, 0, stream>>>(h, out_w, out_b, (float*)d_out);
}
